// L2SelfAttention_41644002902524
// MI455X (gfx1250) — matrix operand order repaired and checked by reading, not compiled, not run on hardware
//
#include <hip/hip_runtime.h>
#include <hip/hip_bf16.h>

typedef __attribute__((ext_vector_type(16))) _Float16 v16h;
typedef __attribute__((ext_vector_type(8)))  _Float16 v8h;
typedef __attribute__((ext_vector_type(8)))  float    v8f;

#define S_LEN 4096
#define D_DIM 512
#define B_CNT 4
#define NROWS (B_CNT * S_LEN)   // 16384
#define KC    256               // keys per chunk in flash kernel

__device__ __forceinline__ v8f wmma_f16(v16h a, v16h b, v8f c) {
    return __builtin_amdgcn_wmma_f32_16x16x32_f16(
        /*neg_a=*/false, a, /*neg_b=*/false, b,
        /*c_mod=*/(short)0, c, /*reuse_a=*/false, /*reuse_b=*/false);
}

// A-fragment (16-bit 16x32): lane (m=lane&15, h=lane>>4) holds
// elems 0..15 -> K = 16h+0..15.
__device__ __forceinline__ v16h loadA(const _Float16* rowbase, int h) {
    v8h lo = *(const v8h*)(rowbase + 16 * h);
    v8h hi = *(const v8h*)(rowbase + 16 * h + 8);
    return __builtin_shufflevector(lo, hi, 0,1,2,3,4,5,6,7,8,9,10,11,12,13,14,15);
}

// ---------------------------------------------------------------- convert
// 8 elements per thread: 2x float4 in, 1x 16B half-vector out.
__global__ void cvt_f32_f16(const float* __restrict__ in,
                            _Float16* __restrict__ out, int n8) {
    int i = blockIdx.x * blockDim.x + threadIdx.x;
    if (i < n8) {
        const float4* src = (const float4*)in + 2 * (size_t)i;
        float4 a = src[0], b = src[1];
        v8h o;
        o[0] = (_Float16)a.x; o[1] = (_Float16)a.y;
        o[2] = (_Float16)a.z; o[3] = (_Float16)a.w;
        o[4] = (_Float16)b.x; o[5] = (_Float16)b.y;
        o[6] = (_Float16)b.z; o[7] = (_Float16)b.w;
        ((v8h*)out)[i] = o;
    }
}

// ---------------------------------------------------------------- QKV GEMM
// Q[m,e] = sum_d x[m,d] * W[e,d]  (torch Linear: W is [out,in])
// B-fragment (32x16): lane (n=lane&15,h=lane>>4), elem e -> K = 16h+e.
__global__ void __launch_bounds__(256)
qkv_proj_kernel(const _Float16* __restrict__ xh,
                const _Float16* __restrict__ Wq,
                const _Float16* __restrict__ Wk,
                const _Float16* __restrict__ Wv,
                _Float16* __restrict__ Qh,
                _Float16* __restrict__ Kh,
                _Float16* __restrict__ Vt) {
    const int w    = threadIdx.x >> 5;
    const int lane = threadIdx.x & 31;
    const int col  = lane & 15;
    const int h    = lane >> 4;

    const int gw      = blockIdx.x * 8 + w;   // 32768 wave-tiles
    const int rowTile = gw >> 5;              // 1024 row tiles of 16
    const int colTile = gw & 31;              // 32 col tiles of 16
    const int m0 = rowTile * 16;
    const int n0 = colTile * 16;

    v8f accq = {}, acck = {}, accv = {};
    const _Float16* xrow = xh + (size_t)(m0 + col) * D_DIM;
    const _Float16* wqr  = Wq + (size_t)(n0 + col) * D_DIM + 16 * h;
    const _Float16* wkr  = Wk + (size_t)(n0 + col) * D_DIM + 16 * h;
    const _Float16* wvr  = Wv + (size_t)(n0 + col) * D_DIM + 16 * h;

#pragma unroll 4
    for (int dc = 0; dc < 16; ++dc) {
        v16h a  = loadA(xrow + dc * 32, h);
        v16h bq = *(const v16h*)(wqr + dc * 32);
        v16h bk = *(const v16h*)(wkr + dc * 32);
        v16h bv = *(const v16h*)(wvr + dc * 32);
        accq = wmma_f16(a, bq, accq);
        acck = wmma_f16(a, bk, acck);
        accv = wmma_f16(a, bv, accv);
    }

    // C layout: M = r + 8h, N = col
    const int b    = m0 / S_LEN;
    const int rloc = m0 % S_LEN;
#pragma unroll
    for (int r = 0; r < 8; ++r) {
        int row = m0 + r + 8 * h;
        Qh[(size_t)row * D_DIM + n0 + col] = (_Float16)accq[r];
        Kh[(size_t)row * D_DIM + n0 + col] = (_Float16)acck[r];
        // V transposed per batch: Vt[b][dim][key]
        Vt[(size_t)b * D_DIM * S_LEN + (size_t)(n0 + col) * S_LEN +
           rloc + r + 8 * h] = (_Float16)accv[r];
    }
}

// ---------------------------------------------------------------- |K|^2
__global__ void knorm_kernel(const _Float16* __restrict__ Kh,
                             float* __restrict__ kn) {
    int r = blockIdx.x * blockDim.x + threadIdx.x;   // 16384 rows
    const v8h* row = (const v8h*)(Kh + (size_t)r * D_DIM);
    float s = 0.f;
#pragma unroll 4
    for (int j = 0; j < D_DIM / 8; ++j) {
        v8h v = row[j];
#pragma unroll
        for (int e = 0; e < 8; ++e) { float f = (float)v[e]; s += f * f; }
    }
    kn[r] = s;
}

// ---------------------------------------------------------------- flash attn
// Block: 256 thr / 8 waves; owns 32 query rows; loops 256-key chunks.
// wave w: qsub = w>>2 ; group g = w&3 ;
//   scores: 4 key-subtiles {g, g+4, g+8, g+12} (one shared A fragment)
//   PV:     dim range g*128 .. +127 (8 dim tiles of 16).
__global__ void __launch_bounds__(256)
flash_l2_attn_kernel(const _Float16* __restrict__ Qh,
                     const _Float16* __restrict__ Kh,
                     const _Float16* __restrict__ Vt,
                     const float* __restrict__ knorm,
                     float* __restrict__ out) {
    __shared__ _Float16 Qs[32 * D_DIM];   // 32 KB
    __shared__ _Float16 Ps[32 * KC];      // 16 KB
    __shared__ float redmax[2][4][16];
    __shared__ float redsum[2][4][16];
    __shared__ float mrow[32], lrow[32], frow[32];

    const int w    = threadIdx.x >> 5;
    const int lane = threadIdx.x & 31;
    const int col  = lane & 15;
    const int h    = lane >> 4;
    const int qsub = w >> 2;
    const int g    = w & 3;

    const int b     = blockIdx.x / (S_LEN / 32);
    const int qbase = (blockIdx.x % (S_LEN / 32)) * 32;

    // Stage 32 x 512 Q tile into LDS, 16B vector copies.
    {
        const float4* src = (const float4*)(Qh + (size_t)(b * S_LEN + qbase) * D_DIM);
        float4* dst = (float4*)Qs;
#pragma unroll
        for (int i = threadIdx.x; i < 32 * D_DIM / 8; i += 256) dst[i] = src[i];
    }
    if (threadIdx.x < 32) { mrow[threadIdx.x] = -1e30f; lrow[threadIdx.x] = 0.f; }
    __syncthreads();

    v8f acc[8];
#pragma unroll
    for (int t = 0; t < 8; ++t) acc[t] = (v8f){};

    const _Float16* Kb = Kh + (size_t)b * S_LEN * D_DIM;
    const _Float16* Vb = Vt + (size_t)b * D_DIM * S_LEN;
    const _Float16* qrow = Qs + (qsub * 16 + col) * D_DIM;

    for (int kc = 0; kc < S_LEN / KC; ++kc) {
        const int kbase = kc * KC;

        // ---- scores: four 16x16 tiles per wave sharing one A fragment
        v8f sc[4];
        const _Float16* kr[4];
        int key[4];
#pragma unroll
        for (int j = 0; j < 4; ++j) {
            sc[j]  = (v8f){};
            key[j] = kbase + (g + 4 * j) * 16 + col;
            kr[j]  = Kb + (size_t)key[j] * D_DIM + 16 * h;
        }
        // prefetch next chunk's K rows (hint; global_prefetch_b8)
        if (kc + 1 < S_LEN / KC) {
#pragma unroll
            for (int j = 0; j < 4; ++j)
                __builtin_prefetch(kr[j] + (size_t)KC * D_DIM, 0, 0);
        }
#pragma unroll 4
        for (int dc = 0; dc < 16; ++dc) {
            v16h a = loadA(qrow + dc * 32, h);
            v16h b0 = *(const v16h*)(kr[0] + dc * 32);
            v16h b1 = *(const v16h*)(kr[1] + dc * 32);
            v16h b2 = *(const v16h*)(kr[2] + dc * 32);
            v16h b3 = *(const v16h*)(kr[3] + dc * 32);
            sc[0] = wmma_f16(a, b0, sc[0]);
            sc[1] = wmma_f16(a, b1, sc[1]);
            sc[2] = wmma_f16(a, b2, sc[2]);
            sc[3] = wmma_f16(a, b3, sc[3]);
        }
        float pm[8];
#pragma unroll
        for (int j = 0; j < 4; ++j) {
            float knj = knorm[(size_t)b * S_LEN + key[j]];
#pragma unroll
            for (int r = 0; r < 8; ++r) {
                sc[j][r] = 2.f * sc[j][r] - knj;
                pm[r] = (j == 0) ? sc[0][r] : fmaxf(pm[r], sc[j][r]);
            }
        }
        // row-max across 16 columns (stays inside each 16-lane half)
#pragma unroll
        for (int mask = 1; mask < 16; mask <<= 1)
#pragma unroll
            for (int r = 0; r < 8; ++r)
                pm[r] = fmaxf(pm[r], __shfl_xor(pm[r], mask, 32));
        if (col == 0)
#pragma unroll
            for (int r = 0; r < 8; ++r) redmax[qsub][g][r + 8 * h] = pm[r];
        __syncthreads();

        if (w == 0) {   // online-max update, one lane per query row
            int row = lane, qs = row >> 4, rr = row & 15;
            float cm = fmaxf(fmaxf(redmax[qs][0][rr], redmax[qs][1][rr]),
                             fmaxf(redmax[qs][2][rr], redmax[qs][3][rr]));
            float mold = mrow[row];
            float mnew = fmaxf(mold, cm);
            mrow[row] = mnew;
            frow[row] = __expf(mold - mnew);
        }
        __syncthreads();

        // ---- P = exp(s - m), write to LDS, partial row sums
        float ps[8];
#pragma unroll
        for (int r = 0; r < 8; ++r) {
            int row = qsub * 16 + r + 8 * h;
            float mn = mrow[row];
            ps[r] = 0.f;
#pragma unroll
            for (int j = 0; j < 4; ++j) {
                float p = __expf(sc[j][r] - mn);
                ps[r] += p;
                Ps[row * KC + (g + 4 * j) * 16 + col] = (_Float16)p;
            }
        }
#pragma unroll
        for (int mask = 1; mask < 16; mask <<= 1)
#pragma unroll
            for (int r = 0; r < 8; ++r) ps[r] += __shfl_xor(ps[r], mask, 32);
        if (col == 0)
#pragma unroll
            for (int r = 0; r < 8; ++r) redsum[qsub][g][r + 8 * h] = ps[r];
        __syncthreads();

        if (w == 0) {
            int row = lane, qs = row >> 4, rr = row & 15;
            float cs = redsum[qs][0][rr] + redsum[qs][1][rr] +
                       redsum[qs][2][rr] + redsum[qs][3][rr];
            lrow[row] = lrow[row] * frow[row] + cs;
        }
        __syncthreads();

        // ---- rescale accumulators, then P·V over this 256-key chunk
        float fr[8];
#pragma unroll
        for (int r = 0; r < 8; ++r) fr[r] = frow[qsub * 16 + r + 8 * h];
#pragma unroll
        for (int t = 0; t < 8; ++t)
#pragma unroll
            for (int r = 0; r < 8; ++r) acc[t][r] *= fr[r];

        const _Float16* prow = Ps + (qsub * 16 + col) * KC;
#pragma unroll 2
        for (int kk = 0; kk < KC / 32; ++kk) {
            v16h a = loadA(prow + kk * 32, h);     // P as A-fragment
#pragma unroll
            for (int t = 0; t < 8; ++t) {
                int dim = g * 128 + t * 16 + col;  // B lane column = dim
                v16h bv = *(const v16h*)(Vb + (size_t)dim * S_LEN +
                                         kbase + kk * 32 + 16 * h);
                acc[t] = wmma_f16(a, bv, acc[t]);
            }
        }
        // next iteration's LDS writes are all fenced by its own barriers
    }

    // ---- epilogue: divide by l, store f32 [B,S,D]
    float li[8];
#pragma unroll
    for (int r = 0; r < 8; ++r) li[r] = 1.f / lrow[qsub * 16 + r + 8 * h];
    float* Og = out + ((size_t)b * S_LEN + qbase + qsub * 16) * D_DIM;
#pragma unroll
    for (int t = 0; t < 8; ++t)
#pragma unroll
        for (int r = 0; r < 8; ++r)
            Og[(size_t)(r + 8 * h) * D_DIM + g * 128 + t * 16 + col] =
                acc[t][r] * li[r];
}

// ---------------------------------------------------------------- launch
extern "C" void kernel_launch(void* const* d_in, const int* in_sizes, int n_in,
                              void* d_out, int out_size, void* d_ws, size_t ws_size,
                              hipStream_t stream) {
    (void)in_sizes; (void)n_in; (void)out_size; (void)ws_size;
    const float* x  = (const float*)d_in[0];
    const float* Wq = (const float*)d_in[1];
    const float* Wk = (const float*)d_in[2];
    const float* Wv = (const float*)d_in[3];
    float* out = (float*)d_out;

    const int NX = NROWS * D_DIM;          // 8,388,608
    const int NW = D_DIM * D_DIM;          // 262,144

    uint8_t* ws = (uint8_t*)d_ws;
    size_t off = 0;
    _Float16* xh  = (_Float16*)(ws + off); off += (size_t)NX * 2;
    _Float16* Wqh = (_Float16*)(ws + off); off += (size_t)NW * 2;
    _Float16* Wkh = (_Float16*)(ws + off); off += (size_t)NW * 2;
    _Float16* Wvh = (_Float16*)(ws + off); off += (size_t)NW * 2;
    _Float16* Qh  = (_Float16*)(ws + off); off += (size_t)NX * 2;
    _Float16* Kh  = (_Float16*)(ws + off); off += (size_t)NX * 2;
    _Float16* Vt  = (_Float16*)(ws + off); off += (size_t)NX * 2;
    float*    kn  = (float*)(ws + off);    off += (size_t)NROWS * 4;

    cvt_f32_f16<<<(NX / 8 + 255) / 256, 256, 0, stream>>>(x,  xh,  NX / 8);
    cvt_f32_f16<<<(NW / 8 + 255) / 256, 256, 0, stream>>>(Wq, Wqh, NW / 8);
    cvt_f32_f16<<<(NW / 8 + 255) / 256, 256, 0, stream>>>(Wk, Wkh, NW / 8);
    cvt_f32_f16<<<(NW / 8 + 255) / 256, 256, 0, stream>>>(Wv, Wvh, NW / 8);

    // 1024 row-tiles x 32 col-tiles = 32768 wave tiles / 8 waves per block
    qkv_proj_kernel<<<4096, 256, 0, stream>>>(xh, Wqh, Wkh, Wvh, Qh, Kh, Vt);

    knorm_kernel<<<NROWS / 256, 256, 0, stream>>>(Kh, kn);

    // B * S/32 = 512 blocks, 8 waves each
    flash_l2_attn_kernel<<<B_CNT * (S_LEN / 32), 256, 0, stream>>>(Qh, Kh, Vt, kn, out);
}
